// Decoder_79207786872969
// MI455X (gfx1250) — compile-verified, matrix operand order checked by reading, not run on hardware
//
#include <hip/hip_runtime.h>
#include <hip/hip_bf16.h>

// ---------------------------------------------------------------------------
// 2-layer LSTM scan, B=1024, H=1024, T=256, fused WMMA bf16 GEMM + cell.
// B fragments software-pipelined (distance 3); slab loop unrolled x2 with
// statically double-buffered A fragments (no cross-slab register copies).
// ---------------------------------------------------------------------------

typedef __attribute__((ext_vector_type(16))) __bf16 v16bf;
typedef __attribute__((ext_vector_type(8)))  __bf16 v8bf;
typedef __attribute__((ext_vector_type(8)))  float  v8f;

#define B_   1024
#define H_   1024
#define T_   256
#define G4H_ 4096

// ---- f32 -> bf16 row converter (also used to stack [W_ih1 | W_hh1]) -------
__global__ void k_cvt_rows(const float* __restrict__ src, __bf16* __restrict__ dst,
                           int rows, int cols, int dstStride, int dstOff) {
  int idx = blockIdx.x * blockDim.x + threadIdx.x;
  int n = rows * cols;
  if (idx >= n) return;
  int r = idx / cols;
  int c = idx - r * cols;
  dst[(size_t)r * dstStride + dstOff + c] = (__bf16)src[idx];
}

__device__ __forceinline__ float sigm(float x) {
  return 1.0f / (1.0f + __expf(-x));
}

__device__ __forceinline__ v16bf load_frag(const __bf16* __restrict__ p) {
  v8bf lo = *(const v8bf*)p;
  v8bf hi = *(const v8bf*)(p + 16);
  return __builtin_shufflevector(lo, hi,
      0,1,2,3,4,5,6,7,8,9,10,11,12,13,14,15);
}

// ---- fused LSTM layer: gates = A @ W^T (+ x*W_ih0 + biases), cell update --
// A (bf16) is [A0 | A1] along K (A1 unused when Kdim==H).
// W (bf16) is (4H x Kdim) row-major (row = gate output column).
// 128-thread block = 64(batch) x 64(j) tile; wave w owns rows [rb+16w, +16)
// with all four gates resident in WMMA accumulators; cell epilogue in regs.
// Requires Kdim % 64 == 0 (slab loop unrolled by 2).
__global__ __launch_bounds__(128)
void k_lstm_layer(const __bf16* __restrict__ A0,
                  const __bf16* __restrict__ A1,
                  const __bf16* __restrict__ W,
                  int Kdim,
                  const float* __restrict__ b_ih,
                  const float* __restrict__ b_hh,
                  const float* __restrict__ xin,   // B floats (layer 0) or null
                  const float* __restrict__ Wih0,  // 4H floats (layer 0) or null
                  float* __restrict__ c_state,     // B x H, updated in place
                  __bf16* __restrict__ h_out)      // B x H bf16
{
  const int lane   = threadIdx.x & 31;
  const int wave   = threadIdx.x >> 5;   // 0..3 : row sub-tile
  const int laneLo = lane & 15;
  const int grp    = lane >> 4;          // 0/1 : K-half select per ISA layout
  const int rb = blockIdx.x * 64;
  const int jb = blockIdx.y * 64;

  v8f acc[16];                           // acc[f], f = g*4 + jt
  const v8f vzero = {0.f,0.f,0.f,0.f,0.f,0.f,0.f,0.f};
#pragma unroll
  for (int f = 0; f < 16; ++f) acc[f] = vzero;

  const int rowA = rb + wave * 16 + laneLo;

  // Fragment address helpers.
  auto aptr = [&](int kk) -> const __bf16* {
    const __bf16* base;
    int ko;
    if (kk < H_) { base = A0; ko = kk; } else { base = A1; ko = kk - H_; }
    return base + (size_t)rowA * H_ + ko + grp * 8;
  };
  auto bptr = [&](int kk, int f) -> const __bf16* {
    const int wrow = (f >> 2) * H_ + jb + (f & 3) * 16 + laneLo; // W row = gate col
    return W + (size_t)wrow * Kdim + kk + grp * 8;
  };

  const int nslab = Kdim >> 5;           // 32 or 64 (even)

  // Pipeline prologue: A for slab 0, B fragments 0..2 in flight.
  v16bf aEven = load_frag(aptr(0));
  v16bf aOdd;
  v16bf bring[4];
  bring[0] = load_frag(bptr(0, 0));
  bring[1] = load_frag(bptr(0, 1));
  bring[2] = load_frag(bptr(0, 2));

  for (int s = 0; s < nslab; s += 2) {
    const int kk0 = s * 32;
    const int kk1 = kk0 + 32;
    const int kk2 = kk0 + 64;

    // ---- even slab: consumes aEven; slab s+1 always exists ----
    aOdd = load_frag(aptr(kk1));
#pragma unroll
    for (int f = 0; f < 16; ++f) {
      if (f < 13) bring[(f + 3) & 3] = load_frag(bptr(kk0, f + 3));
      else        bring[(f + 3) & 3] = load_frag(bptr(kk1, f - 13));
      acc[f] = __builtin_amdgcn_wmma_f32_16x16x32_bf16(
          false, aEven, false, bring[f & 3], (short)0, acc[f], false, false);
    }

    // ---- odd slab: consumes aOdd; prefetch slab s+2 if it exists ----
    const bool more = (s + 2 < nslab);
    if (more) aEven = load_frag(aptr(kk2));
#pragma unroll
    for (int f = 0; f < 16; ++f) {
      if (f < 13)    bring[(f + 3) & 3] = load_frag(bptr(kk1, f + 3));
      else if (more) bring[(f + 3) & 3] = load_frag(bptr(kk2, f - 13));
      acc[f] = __builtin_amdgcn_wmma_f32_16x16x32_bf16(
          false, aOdd, false, bring[f & 3], (short)0, acc[f], false, false);
    }
  }

  // Epilogue: biases + rank-1 input term + LSTM cell, all in C-layout regs.
#pragma unroll
  for (int jt = 0; jt < 4; ++jt) {
    const int j = jb + jt * 16 + laneLo;    // column within H
    float bs[4], wx[4];
#pragma unroll
    for (int g = 0; g < 4; ++g) {
      bs[g] = b_ih[g * H_ + j] + b_hh[g * H_ + j];
      wx[g] = Wih0 ? Wih0[g * H_ + j] : 0.0f;
    }
#pragma unroll
    for (int v = 0; v < 8; ++v) {
      const int row = rb + wave * 16 + grp * 8 + v;  // C/D layout: M = v + grp*8
      const float xv = xin ? xin[row] : 0.0f;
      float gi = acc[0 * 4 + jt][v] + bs[0] + xv * wx[0];
      float gf = acc[1 * 4 + jt][v] + bs[1] + xv * wx[1];
      float gg = acc[2 * 4 + jt][v] + bs[2] + xv * wx[2];
      float go = acc[3 * 4 + jt][v] + bs[3] + xv * wx[3];
      const size_t off = (size_t)row * H_ + j;
      float c_old = c_state[off];
      float c_new = sigm(gf) * c_old + sigm(gi) * tanhf(gg);
      float h_new = sigm(go) * tanhf(c_new);
      c_state[off] = c_new;
      h_out[off] = (__bf16)h_new;
    }
  }
}

// ---- output GEMV + loss + feedback: one wave per batch row ----------------
__global__ __launch_bounds__(256)
void k_out_gemv(const __bf16* __restrict__ hb, const float* __restrict__ Wout,
                const float* __restrict__ bout, const float* __restrict__ targets,
                int t, float* __restrict__ preds, float* __restrict__ loss,
                float* __restrict__ xnext)
{
  const int gwave = (int)((blockIdx.x * blockDim.x + threadIdx.x) >> 5);
  const int lane  = threadIdx.x & 31;
  if (gwave >= B_) return;
  const __bf16* hp = hb + (size_t)gwave * H_;
  float s = 0.0f;
#pragma unroll 4
  for (int k = lane; k < H_; k += 32) s += (float)hp[k] * Wout[k];
#pragma unroll
  for (int m = 16; m >= 1; m >>= 1) s += __shfl_xor(s, m, 32);
  if (lane == 0) {
    float out = s + bout[0];
    preds[(size_t)gwave * T_ + t] = out;
    xnext[gwave] = out;
    float e = out - targets[(size_t)gwave * T_ + t];
    atomicAdd(loss, e * e * (1.0f / (float)B_));
  }
}

// ---------------------------------------------------------------------------
extern "C" void kernel_launch(void* const* d_in, const int* in_sizes, int n_in,
                              void* d_out, int out_size, void* d_ws, size_t ws_size,
                              hipStream_t stream) {
  (void)in_sizes; (void)n_in; (void)out_size; (void)ws_size;
  const float* outputs = (const float*)d_in[0];   // (B, T)
  const float* h0      = (const float*)d_in[1];   // (2, B, H)
  const float* c0      = (const float*)d_in[2];   // (2, B, H)
  const float* W_ih0   = (const float*)d_in[3];   // (4H, 1)
  const float* W_hh0   = (const float*)d_in[4];   // (4H, H)
  const float* b_ih0   = (const float*)d_in[5];
  const float* b_hh0   = (const float*)d_in[6];
  const float* W_ih1   = (const float*)d_in[7];   // (4H, H)
  const float* W_hh1   = (const float*)d_in[8];   // (4H, H)
  const float* b_ih1   = (const float*)d_in[9];
  const float* b_hh1   = (const float*)d_in[10];
  const float* W_out   = (const float*)d_in[11];  // (1, H)
  const float* b_out   = (const float*)d_in[12];

  float* preds = (float*)d_out;                   // B*T
  float* loss  = preds + (size_t)B_ * T_;         // + scalar loss

  // ---- workspace carve-out ----
  char* ws = (char*)d_ws;
  size_t off = 0;
  auto alloc = [&](size_t bytes) -> void* {
    off = (off + 255) & ~(size_t)255;
    void* p = ws + off;
    off += bytes;
    return p;
  };
  __bf16* wB0   = (__bf16*)alloc((size_t)G4H_ * H_ * 2);        // 8 MB
  __bf16* wB1   = (__bf16*)alloc((size_t)G4H_ * 2 * H_ * 2);    // 16 MB
  __bf16* hA[2] = { (__bf16*)alloc((size_t)B_ * H_ * 2),
                    (__bf16*)alloc((size_t)B_ * H_ * 2) };
  __bf16* hB[2] = { (__bf16*)alloc((size_t)B_ * H_ * 2),
                    (__bf16*)alloc((size_t)B_ * H_ * 2) };
  float*  cA    = (float*)alloc((size_t)B_ * H_ * 4);
  float*  cB    = (float*)alloc((size_t)B_ * H_ * 4);
  float*  xbuf  = (float*)alloc((size_t)B_ * 4);

  // ---- one-time prep (every call; deterministic) ----
  {
    const int thr = 256;
    int nW = G4H_ * H_;
    int nH = B_ * H_;
    k_cvt_rows<<<(nW + thr - 1) / thr, thr, 0, stream>>>(W_hh0, wB0, G4H_, H_, H_, 0);
    k_cvt_rows<<<(nW + thr - 1) / thr, thr, 0, stream>>>(W_ih1, wB1, G4H_, H_, 2 * H_, 0);
    k_cvt_rows<<<(nW + thr - 1) / thr, thr, 0, stream>>>(W_hh1, wB1, G4H_, H_, 2 * H_, H_);
    k_cvt_rows<<<(nH + thr - 1) / thr, thr, 0, stream>>>(h0,            hA[0], B_, H_, H_, 0);
    k_cvt_rows<<<(nH + thr - 1) / thr, thr, 0, stream>>>(h0 + (size_t)B_ * H_, hB[0], B_, H_, H_, 0);
    hipMemcpyAsync(cA, c0,                     (size_t)B_ * H_ * 4, hipMemcpyDeviceToDevice, stream);
    hipMemcpyAsync(cB, c0 + (size_t)B_ * H_,   (size_t)B_ * H_ * 4, hipMemcpyDeviceToDevice, stream);
    hipMemsetAsync(xbuf, 0, (size_t)B_ * 4, stream);
    hipMemsetAsync(loss, 0, sizeof(float), stream);
  }

  const dim3 lgrid(B_ / 64, H_ / 64);   // 16 x 16 blocks
  const dim3 lblk(128);

  for (int t = 0; t < T_; ++t) {
    const int cur = t & 1, nxt = cur ^ 1;
    // layer 0: gates = hA@W_hh0^T + x*W_ih0 + biases
    k_lstm_layer<<<lgrid, lblk, 0, stream>>>(
        hA[cur], (const __bf16*)nullptr, wB0, H_,
        b_ih0, b_hh0, xbuf, W_ih0, cA, hA[nxt]);
    // layer 1: gates = [hA_new | hB]@[W_ih1 | W_hh1]^T + biases
    k_lstm_layer<<<lgrid, lblk, 0, stream>>>(
        hA[nxt], hB[cur], wB1, 2 * H_,
        b_ih1, b_hh1, (const float*)nullptr, (const float*)nullptr, cB, hB[nxt]);
    // output head + loss + feedback
    k_out_gemv<<<B_ / 8, 256, 0, stream>>>(
        hB[nxt], W_out, b_out, outputs, t, preds, loss, xbuf);
  }
}
